// CrossTransformer_40939628265561
// MI455X (gfx1250) — compile-verified, matrix operand order checked by reading, not run on hardware
//
#include <hip/hip_runtime.h>

typedef unsigned short u16;
typedef unsigned int   u32;
typedef __attribute__((ext_vector_type(16))) __bf16 v16bf;
typedef __attribute__((ext_vector_type(8)))  float  v8f;

#define BATCH 64
#define NHEAD 16
#define DHEAD 64
#define INNER 1024
#define NKV   2048   // 2*INNER columns of the KV projection
#define EPSLN 1e-5f

__device__ __forceinline__ u16 f2bf(float f) {
    u32 u = __float_as_uint(f);
    u32 r = u + 0x7FFFu + ((u >> 16) & 1u);   // round-to-nearest-even
    return (u16)(r >> 16);
}
__device__ __forceinline__ float bf2f(u16 h) {
    return __uint_as_float(((u32)h) << 16);
}

union Frag { v16bf v; u32 d[8]; };

// ---------------------------------------------------------------------------
// Copy full input token tensors into d_out (patches final; cls rows updated
// in place by the layer loop).
// ---------------------------------------------------------------------------
__global__ __launch_bounds__(256) void copy_out_kern(const float* __restrict__ sm,
                                                     const float* __restrict__ lg,
                                                     float* __restrict__ out,
                                                     long sm_total) {
    long i = (long)blockIdx.x * 256 + threadIdx.x;
    out[i] = (i < sm_total) ? sm[i] : lg[i - sm_total];
}

// ---------------------------------------------------------------------------
// Pack patch tokens to bf16 row-major A[b*L + 1 + t][k]; row b*L is reserved
// for the per-layer normalized cls token.
// ---------------------------------------------------------------------------
__global__ __launch_bounds__(256) void pack_patch_kern(const float* __restrict__ tok,
                                                       u16* __restrict__ A,
                                                       int Np, int K, int L) {
    long idx = (long)blockIdx.x * 256 + threadIdx.x;
    int  k   = (int)(idx % K);
    long r   = idx / K;
    int  t   = (int)(r % Np);
    int  b   = (int)(r / Np);
    A[((size_t)b * L + 1 + t) * K + k] =
        f2bf(tok[((size_t)b * (Np + 1) + 1 + t) * K + k]);
}

// ---------------------------------------------------------------------------
// Pack wkv (D,K,2048) f32 into WMMA B-fragment order:
//   Wp[layer][kt][nt][lane][e]  -- 16 contiguous bf16 per lane (32B).
// B-operand layout (ISA 7.12.2): lane n=L%16, h=L/16; dword j holds
// K = (j<4 ? 2j : 16+2(j-4)) + 8h  and K+1, column nt*16+n.
// ---------------------------------------------------------------------------
__global__ __launch_bounds__(256) void pack_w_kern(const float* __restrict__ wsrc,
                                                   u16* __restrict__ Wp, int K) {
    long idx   = (long)blockIdx.x * 256 + threadIdx.x;
    long perL  = (long)K * NKV;
    int  layer = (int)(idx / perL);
    long rem   = idx % perL;
    int  e    = (int)(rem & 15);
    int  lane = (int)((rem >> 4) & 31);
    int  nt   = (int)((rem >> 9) & 127);
    int  kt   = (int)(rem >> 16);
    int  h = lane >> 4, n = lane & 15;
    int  j = e >> 1, odd = e & 1;
    int  k0 = ((j < 4) ? (2 * j) : (16 + 2 * (j - 4))) + 8 * h;
    int  k  = kt * 32 + k0 + odd;
    int  col = nt * 16 + n;
    Wp[idx] = f2bf(wsrc[((size_t)layer * K + k) * NKV + col]);
}

// ---------------------------------------------------------------------------
// cls pre-stage: x = cls @ pin_w + pin_b ; LayerNorm -> xn ; xn->bf16 into A
// row b*L ; q = (xn @ wq) * DH^-0.5.  One block per batch.
// ---------------------------------------------------------------------------
__global__ __launch_bounds__(256) void cls_pre_kern(const float* __restrict__ cls_base,
                                                    long cls_stride,
                                                    const float* __restrict__ pin_w,
                                                    const float* __restrict__ pin_b,
                                                    const float* __restrict__ ng,
                                                    const float* __restrict__ nb,
                                                    int din, int dout,
                                                    const float* __restrict__ wq,
                                                    u16* __restrict__ A, int L,
                                                    float* __restrict__ qb) {
    __shared__ float scls[1024];
    __shared__ float sx[1024];
    __shared__ float sr[16];
    int tid = threadIdx.x, lane = tid & 31, w = tid >> 5;
    int b = blockIdx.x;
    const float* cls = cls_base + (size_t)b * cls_stride;

    for (int k = tid; k < din; k += 256) scls[k] = cls[k];
    __syncthreads();

    for (int e = tid; e < dout; e += 256) {
        float acc = pin_b[e];
        for (int k = 0; k < din; ++k) acc += scls[k] * pin_w[(size_t)k * dout + e];
        sx[e] = acc;
    }
    __syncthreads();

    float lsum = 0.f, lsq = 0.f;
    for (int e = tid; e < dout; e += 256) { float v = sx[e]; lsum += v; lsq += v * v; }
    for (int o = 16; o > 0; o >>= 1) {
        lsum += __shfl_xor(lsum, o, 32);
        lsq  += __shfl_xor(lsq,  o, 32);
    }
    if (lane == 0) { sr[w] = lsum; sr[8 + w] = lsq; }
    __syncthreads();
    if (tid == 0) {
        float a = 0.f, s2 = 0.f;
        for (int i = 0; i < 8; ++i) { a += sr[i]; s2 += sr[8 + i]; }
        float mu = a / dout;
        sr[0] = mu;
        sr[1] = s2 / dout - mu * mu;
    }
    __syncthreads();
    float mu = sr[0];
    float rs = rsqrtf(sr[1] + EPSLN);
    for (int e = tid; e < dout; e += 256) {
        float xn = (sx[e] - mu) * rs * ng[e] + nb[e];
        sx[e] = xn;
        A[(size_t)b * L * dout + e] = f2bf(xn);   // row 0 of this batch block (K == dout)
    }
    __syncthreads();

    const float scale = 0.125f;   // DH^-0.5
    for (int jq = tid; jq < INNER; jq += 256) {
        float acc = 0.f;
        for (int e = 0; e < dout; ++e) acc += sx[e] * wq[(size_t)e * INNER + jq];
        qb[(size_t)b * INNER + jq] = acc * scale;
    }
}

// ---------------------------------------------------------------------------
// KV projection GEMM: KV[M x 2048] (bf16) = A[M x K] (bf16) @ Wp (packed bf16)
// 256 threads = 8 waves (4 M-tiles x 2 N-strips of 64). v_wmma_f32_16x16x32_bf16.
// M = 64*L is a multiple of 64 -> no bounds checks, EXEC all ones.
// ---------------------------------------------------------------------------
__global__ __launch_bounds__(256) void kv_gemm_kern(const u16* __restrict__ A,
                                                    const u16* __restrict__ Wp,
                                                    u16* __restrict__ KV, int K) {
    int tid  = threadIdx.x;
    int lane = tid & 31;
    int w    = tid >> 5;
    int wm = w & 3, wn = w >> 2;
    int mbase = blockIdx.x * 64 + wm * 16;
    int nbase = blockIdx.y * 128 + wn * 64;
    int h = lane >> 4, r = lane & 15;

    v8f acc[4] = {};
    const u16* arow = A + (size_t)(mbase + r) * K;
    int kTiles = K >> 5;

    for (int kt = 0; kt < kTiles; ++kt) {
        Frag af;
#pragma unroll
        for (int j = 0; j < 8; ++j) {
            int k0 = ((j < 4) ? (2 * j) : (16 + 2 * (j - 4))) + 8 * h;
            af.d[j] = *(const u32*)(arow + kt * 32 + k0);
        }
#pragma unroll
        for (int s = 0; s < 4; ++s) {
            int nt = (nbase >> 4) + s;
            const uint4* bp = (const uint4*)(Wp + (((size_t)kt * (NKV / 16) + nt) * 32 + lane) * 16);
            uint4 lo = bp[0], hi = bp[1];
            Frag bfr;
            bfr.d[0] = lo.x; bfr.d[1] = lo.y; bfr.d[2] = lo.z; bfr.d[3] = lo.w;
            bfr.d[4] = hi.x; bfr.d[5] = hi.y; bfr.d[6] = hi.z; bfr.d[7] = hi.w;
            acc[s] = __builtin_amdgcn_wmma_f32_16x16x32_bf16(
                false, af.v, false, bfr.v, (short)0, acc[s], false, false);
        }
    }
#pragma unroll
    for (int s = 0; s < 4; ++s) {
        int col = nbase + s * 16 + r;
#pragma unroll
        for (int vv = 0; vv < 8; ++vv) {
            int row = mbase + h * 8 + vv;
            KV[(size_t)row * NKV + col] = f2bf(acc[s][vv]);
        }
    }
}

// ---------------------------------------------------------------------------
// Attention for the single-query cls token. One wave per (batch, head):
// lane-per-key scores, shuffle softmax, LDS-staged probs, lane-per-dim P.V.
// ---------------------------------------------------------------------------
__global__ __launch_bounds__(256) void attn_kern(const float* __restrict__ qb,
                                                 const u16* __restrict__ KV,
                                                 int L,
                                                 float* __restrict__ ob) {
    __shared__ float sq[8][64];
    __shared__ float sp[8][416];
    int tid = threadIdx.x, lane = tid & 31, w = tid >> 5;
    int idx = blockIdx.x * 8 + w;
    int b = idx >> 4, hh = idx & 15;

    sq[w][lane]      = qb[(size_t)b * INNER + hh * 64 + lane];
    sq[w][lane + 32] = qb[(size_t)b * INNER + hh * 64 + lane + 32];
    __syncthreads();

    float sc[13];
    float smax = -3.0e38f;
#pragma unroll
    for (int i = 0; i < 13; ++i) {
        int t = lane + i * 32;
        if (t < L) {
            const u16* kr = KV + ((size_t)(b * L + t)) * NKV + hh * 64;
            float acc = 0.f;
#pragma unroll
            for (int d = 0; d < 64; ++d) acc += sq[w][d] * bf2f(kr[d]);
            sc[i] = acc;
            smax = fmaxf(smax, acc);
        }
    }
    for (int o = 16; o > 0; o >>= 1) smax = fmaxf(smax, __shfl_xor(smax, o, 32));

    float esum = 0.f;
#pragma unroll
    for (int i = 0; i < 13; ++i) {
        int t = lane + i * 32;
        if (t < L) { float e = __expf(sc[i] - smax); sp[w][t] = e; esum += e; }
    }
    for (int o = 16; o > 0; o >>= 1) esum += __shfl_xor(esum, o, 32);
    float inv = 1.f / esum;
    __syncthreads();

#pragma unroll
    for (int d2 = 0; d2 < 2; ++d2) {
        int d = lane + d2 * 32;
        float acc = 0.f;
        for (int t = 0; t < L; ++t)
            acc += sp[w][t] * bf2f(KV[((size_t)(b * L + t)) * NKV + INNER + hh * 64 + d]);
        ob[(size_t)b * INNER + hh * 64 + d] = acc * inv;
    }
}

// ---------------------------------------------------------------------------
// cls post-stage: t = o @ wo + wob ; cls = t @ pout_w + pout_b + cls (in place).
// ---------------------------------------------------------------------------
__global__ __launch_bounds__(256) void cls_post_kern(const float* __restrict__ ob,
                                                     const float* __restrict__ wo,
                                                     const float* __restrict__ wob,
                                                     const float* __restrict__ pout_w,
                                                     const float* __restrict__ pout_b,
                                                     int dout, int din,
                                                     float* __restrict__ cls_base,
                                                     long cls_stride) {
    __shared__ float so[1024];
    __shared__ float st[1024];
    int tid = threadIdx.x;
    int b = blockIdx.x;
    float* cls = cls_base + (size_t)b * cls_stride;

    for (int jo = tid; jo < INNER; jo += 256) so[jo] = ob[(size_t)b * INNER + jo];
    __syncthreads();

    for (int e = tid; e < dout; e += 256) {
        float acc = wob[e];
        for (int j = 0; j < INNER; ++j) acc += so[j] * wo[(size_t)j * dout + e];
        st[e] = acc;
    }
    __syncthreads();

    for (int i = tid; i < din; i += 256) {
        float acc = pout_b[i] + cls[i];
        for (int e = 0; e < dout; ++e) acc += st[e] * pout_w[(size_t)e * din + i];
        cls[i] = acc;
    }
}

// ---------------------------------------------------------------------------
extern "C" void kernel_launch(void* const* d_in, const int* in_sizes, int n_in,
                              void* d_out, int out_size, void* d_ws, size_t ws_size,
                              hipStream_t stream) {
    const float* sm_tokens = (const float*)d_in[0];
    const float* lg_tokens = (const float*)d_in[1];
    const float* a_pin_w = (const float*)d_in[2];
    const float* a_pin_b = (const float*)d_in[3];
    const float* a_ng    = (const float*)d_in[4];
    const float* a_nb    = (const float*)d_in[5];
    const float* a_wq    = (const float*)d_in[6];
    const float* a_wkv   = (const float*)d_in[7];
    const float* a_wo    = (const float*)d_in[8];
    const float* a_wob   = (const float*)d_in[9];
    const float* a_pout_w = (const float*)d_in[10];
    const float* a_pout_b = (const float*)d_in[11];
    const float* b_pin_w = (const float*)d_in[12];
    const float* b_pin_b = (const float*)d_in[13];
    const float* b_ng    = (const float*)d_in[14];
    const float* b_nb    = (const float*)d_in[15];
    const float* b_wq    = (const float*)d_in[16];
    const float* b_wkv   = (const float*)d_in[17];
    const float* b_wo    = (const float*)d_in[18];
    const float* b_wob   = (const float*)d_in[19];
    const float* b_pout_w = (const float*)d_in[20];
    const float* b_pout_b = (const float*)d_in[21];

    // Workspace carve (~183 MB total).
    size_t off = 0;
    auto carve = [&](size_t bytes) -> char* {
        char* p = (char*)d_ws + off;
        off += (bytes + 255) & ~(size_t)255;
        return p;
    };
    u16* A_lg = (u16*)carve((size_t)BATCH * 197 * 1024 * 2);   // branch-a activations
    u16* A_sm = (u16*)carve((size_t)BATCH * 401 * 512 * 2);    // branch-b activations
    u16* Wpa  = (u16*)carve((size_t)4 * 1024 * NKV * 2);       // packed a_wkv
    u16* Wpb  = (u16*)carve((size_t)4 * 512 * NKV * 2);        // packed b_wkv
    u16* KV   = (u16*)carve((size_t)BATCH * 401 * NKV * 2);    // reused per stage
    float* qb = (float*)carve((size_t)BATCH * INNER * 4);
    float* ob = (float*)carve((size_t)BATCH * INNER * 4);

    const long SMTOT = (long)BATCH * 401 * 512;     // 13,139,968
    const long TOT   = SMTOT + (long)BATCH * 197 * 1024;

    // Prologue: identity copy of tokens; pack patches + weights to bf16.
    copy_out_kern<<<(int)(TOT / 256), 256, 0, stream>>>(sm_tokens, lg_tokens,
                                                        (float*)d_out, SMTOT);
    pack_patch_kern<<<(64 * 196 * 1024) / 256, 256, 0, stream>>>(lg_tokens, A_lg, 196, 1024, 197);
    pack_patch_kern<<<(64 * 400 * 512) / 256, 256, 0, stream>>>(sm_tokens, A_sm, 400, 512, 401);
    pack_w_kern<<<(4 * 1024 * NKV) / 256, 256, 0, stream>>>(a_wkv, Wpa, 1024);
    pack_w_kern<<<(4 * 512 * NKV) / 256, 256, 0, stream>>>(b_wkv, Wpb, 512);

    float* sm_cls = (float*)d_out;             // row 0 of each (401,512) block
    float* lg_cls = (float*)d_out + SMTOT;     // row 0 of each (197,1024) block
    const long sm_stride = 401 * 512;
    const long lg_stride = 197 * 1024;

    for (int layer = 0; layer < 4; ++layer) {
        // ---- branch a: sm_cls attends over lg patches (din=512, dout=1024, L=197, K=1024)
        cls_pre_kern<<<BATCH, 256, 0, stream>>>(
            sm_cls, sm_stride,
            a_pin_w + (size_t)layer * 512 * 1024, a_pin_b + (size_t)layer * 1024,
            a_ng + (size_t)layer * 1024, a_nb + (size_t)layer * 1024,
            512, 1024,
            a_wq + (size_t)layer * 1024 * 1024,
            A_lg, 197, qb);
        kv_gemm_kern<<<dim3(197, 16), 256, 0, stream>>>(
            A_lg, Wpa + (size_t)layer * 1024 * NKV, KV, 1024);
        attn_kern<<<(BATCH * NHEAD) / 8, 256, 0, stream>>>(qb, KV, 197, ob);
        cls_post_kern<<<BATCH, 256, 0, stream>>>(
            ob,
            a_wo + (size_t)layer * 1024 * 1024, a_wob + (size_t)layer * 1024,
            a_pout_w + (size_t)layer * 1024 * 512, a_pout_b + (size_t)layer * 512,
            1024, 512, sm_cls, sm_stride);

        // ---- branch b: lg_cls attends over sm patches (din=1024, dout=512, L=401, K=512)
        cls_pre_kern<<<BATCH, 256, 0, stream>>>(
            lg_cls, lg_stride,
            b_pin_w + (size_t)layer * 1024 * 512, b_pin_b + (size_t)layer * 512,
            b_ng + (size_t)layer * 512, b_nb + (size_t)layer * 512,
            1024, 512,
            b_wq + (size_t)layer * 512 * 1024,
            A_sm, 401, qb);
        kv_gemm_kern<<<dim3(401, 16), 256, 0, stream>>>(
            A_sm, Wpb + (size_t)layer * 512 * NKV, KV, 512);
        attn_kern<<<(BATCH * NHEAD) / 8, 256, 0, stream>>>(qb, KV, 401, ob);
        cls_post_kern<<<BATCH, 256, 0, stream>>>(
            ob,
            b_wo + (size_t)layer * 1024 * 512, b_wob + (size_t)layer * 512,
            b_pout_w + (size_t)layer * 512 * 1024, b_pout_b + (size_t)layer * 1024,
            512, 1024, lg_cls, lg_stride);
    }
    (void)in_sizes; (void)n_in; (void)out_size; (void)ws_size;
}
